// EncoderBlock_25881472925799
// MI455X (gfx1250) — matrix operand order repaired and checked by reading, not compiled, not run on hardware
//
#include <hip/hip_runtime.h>

#define NRES 4096
#define DMODEL 256
#define KNB 64
#define NHEAD 8
#define PDIM 128

typedef __bf16 bf16_t;
typedef __attribute__((ext_vector_type(16))) __bf16 v16bf;
typedef __attribute__((ext_vector_type(8)))  __bf16 v8bf;
typedef __attribute__((ext_vector_type(8)))  float  v8f;

// ---------------------------------------------------------------------------
// WMMA helpers (wave32, v_wmma_f32_16x16x32_bf16). Layouts per CDNA5 ISA 7.12.2:
//  A 16x32 bf16: lane half h, row m=lane&15, VGPR i holds K = (i/4)*16 + h*8 + (i%4)*2 (+1)
//  B 32x16 bf16: lane half h, col n=lane&15, VGPR i holds K = (i/4)*16 + h*8 + (i%4)*2 (+1)
//  C/D 16x16 f32: VGPR r -> row m = r + 8*h, col n = lane&15
// B matrices are staged in LDS in fragment-major ("swizzled") order so each
// lane's 16 bf16 values are contiguous -> 2x ds_load_b128 per fragment.
// ---------------------------------------------------------------------------
__device__ __forceinline__ v16bf lda_frag(const bf16_t* A, int lda, int k0) {
  const int lane = threadIdx.x & 31;
  const int hh = lane >> 4;
  const int m  = lane & 15;
  v16bf a;
#pragma unroll
  for (int i = 0; i < 8; ++i) {
    int k = k0 + ((i >> 2) << 4) + (hh << 3) + ((i & 3) << 1);
    a[2 * i]     = A[m * lda + k];
    a[2 * i + 1] = A[m * lda + k + 1];
  }
  return a;
}

// Swizzled B: entry index = ((nt * nkt) + kt) * 32 + lane, 16 bf16 each.
__device__ __forceinline__ v16bf ldb_frag_swz(const bf16_t* Bs, int nkt, int kt,
                                              int nt) {
  const int lane = threadIdx.x & 31;
  const bf16_t* p = Bs + (size_t)(((nt * nkt) + kt) * 32 + lane) * 16;
  v8bf lo = *(const v8bf*)p;
  v8bf hi = *(const v8bf*)(p + 8);
  return __builtin_shufflevector(lo, hi, 0, 1, 2, 3, 4, 5, 6, 7, 8, 9, 10, 11,
                                 12, 13, 14, 15);
}

// Stage a K x N weight (getter(k,n) -> fp32) into swizzled bf16 LDS layout.
template <typename F>
__device__ __forceinline__ void stage_swz(bf16_t* Bs, int Kpad, int Nout, F get) {
  const int nkt = Kpad >> 5;
  const int entries = (Nout >> 4) * nkt * 32;
  for (int e = threadIdx.x; e < entries; e += 256) {
    int lane = e & 31;
    int tile = e >> 5;
    int kt = tile % nkt;
    int nt = tile / nkt;
    int hh = lane >> 4, nn = lane & 15;
    int n = nt * 16 + nn;
    int kb = kt * 32 + hh * 8;
    bf16_t* dst = Bs + (size_t)e * 16;
#pragma unroll
    for (int i = 0; i < 8; ++i) {
      dst[2 * i]     = (bf16_t)get(kb + 2 * i + ((i >> 2) << 3), n);
      dst[2 * i + 1] = (bf16_t)get(kb + 2 * i + ((i >> 2) << 3) + 1, n);
    }
  }
}

__device__ __forceinline__ v8f wmma_bf16(v16bf a, v16bf b, v8f c) {
  return __builtin_amdgcn_wmma_f32_16x16x32_bf16(false, a, false, b, (short)0, c,
                                                 false, false);
}

__device__ __forceinline__ float gelu_f(float x) {
  float x3 = x * x * x;
  return 0.5f * x * (1.0f + tanhf(0.7978845608028654f * (x + 0.044715f * x3)));
}

// CDNA5 async global->LDS copy (per-lane 16B), tracked by ASYNCcnt.
__device__ __forceinline__ void async_copy_b128(void* lds_dst, const void* gsrc) {
  unsigned int loff = (unsigned int)(unsigned long long)lds_dst;
  asm volatile("global_load_async_to_lds_b128 %0, %1, off" ::"v"(loff), "v"(gsrc)
               : "memory");
}
__device__ __forceinline__ void async_wait0() {
  asm volatile("s_wait_asynccnt 0x0" ::: "memory");
}

// ---------------------------------------------------------------------------
// Kernel 1: frames (R, ca) + local atom positions (15-dim, padded to 16)
// ---------------------------------------------------------------------------
__global__ __launch_bounds__(256) void frames_kernel(
    const float* __restrict__ pos, float* __restrict__ Rw,
    float* __restrict__ caw, float* __restrict__ lpw) {
  int n = blockIdx.x * blockDim.x + threadIdx.x;
  if (n >= NRES) return;
  const float* p = pos + (size_t)n * 15;
  float nx = p[0], ny = p[1], nz = p[2];
  float cax = p[3], cay = p[4], caz = p[5];
  float cx = p[6], cy = p[7], cz = p[8];

  float d1x = cx - cax, d1y = cy - cay, d1z = cz - caz;
  float i1 = rsqrtf(d1x * d1x + d1y * d1y + d1z * d1z + 1e-12f);
  float e1x = d1x * i1, e1y = d1y * i1, e1z = d1z * i1;

  float v2x = nx - cax, v2y = ny - cay, v2z = nz - caz;
  float dp = v2x * e1x + v2y * e1y + v2z * e1z;
  float p2x = v2x - dp * e1x, p2y = v2y - dp * e1y, p2z = v2z - dp * e1z;
  float i2 = rsqrtf(p2x * p2x + p2y * p2y + p2z * p2z + 1e-12f);
  float e2x = p2x * i2, e2y = p2y * i2, e2z = p2z * i2;

  float e3x = e1y * e2z - e1z * e2y;
  float e3y = e1z * e2x - e1x * e2z;
  float e3z = e1x * e2y - e1y * e2x;

  float* R = Rw + (size_t)n * 9;  // R[d*3+r] = e_r[d]
  R[0] = e1x; R[1] = e2x; R[2] = e3x;
  R[3] = e1y; R[4] = e2y; R[5] = e3y;
  R[6] = e1z; R[7] = e2z; R[8] = e3z;
  caw[n * 3 + 0] = cax; caw[n * 3 + 1] = cay; caw[n * 3 + 2] = caz;

  float* lp = lpw + (size_t)n * 16;
#pragma unroll
  for (int a = 0; a < 5; ++a) {
    float vx = p[a * 3] - cax, vy = p[a * 3 + 1] - cay, vz = p[a * 3 + 2] - caz;
    lp[a * 3 + 0] = e1x * vx + e1y * vy + e1z * vz;
    lp[a * 3 + 1] = e2x * vx + e2y * vy + e2z * vz;
    lp[a * 3 + 2] = e3x * vx + e3y * vy + e3z * vz;
  }
  lp[15] = 0.f;
}

// ---------------------------------------------------------------------------
// Kernel 2: neighbour selection (one block per residue, full O(N) scan in LDS)
// ---------------------------------------------------------------------------
__global__ __launch_bounds__(256) void neighbours_kernel(
    const float* __restrict__ ca, const int* __restrict__ resi,
    const int* __restrict__ chain, const int* __restrict__ batch,
    const unsigned char* __restrict__ mask, int* __restrict__ nb) {
  __shared__ float sdist[NRES];
  __shared__ float swork[NRES];
  __shared__ float rv[256];
  __shared__ int ri[256];

  const float FINF = __builtin_inff();
  int n = blockIdx.x;
  int t = threadIdx.x;
  float cax = ca[n * 3], cay = ca[n * 3 + 1], caz = ca[n * 3 + 2];
  int rn = resi[n], cn = chain[n], bn = batch[n];
  bool mn = mask[n] != 0;

  for (int j = t; j < NRES; j += 256) {
    bool sb = (bn == batch[j]);
    bool sc = (cn == chain[j]);
    bool valid = sb && mn && (mask[j] != 0);
    int dr = rn - resi[j]; if (dr < 0) dr = -dr;
    bool within0 = (dr < 16) && sb && sc;
    float vx = ca[j * 3] - cax, vy = ca[j * 3 + 1] - cay, vz = ca[j * 3 + 2] - caz;
    float d = sqrtf(vx * vx + vy * vy + vz * vz + 1e-12f);
    float de = (!valid || within0) ? FINF : d;
    sdist[j] = de;
    swork[j] = de;
  }
  __syncthreads();

  // cutoff = 16th smallest effective distance
  float cutoff = FINF;
  for (int round = 0; round < 16; ++round) {
    float bv = FINF; int bi = NRES;
    for (int j = t; j < NRES; j += 256) {
      float v = swork[j];
      if (v < bv || (v == bv && j < bi)) { bv = v; bi = j; }
    }
    rv[t] = bv; ri[t] = bi;
    __syncthreads();
    for (int s = 128; s > 0; s >>= 1) {
      if (t < s) {
        if (rv[t + s] < rv[t] || (rv[t + s] == rv[t] && ri[t + s] < ri[t])) {
          rv[t] = rv[t + s]; ri[t] = ri[t + s];
        }
      }
      __syncthreads();
    }
    if (round == 15) cutoff = rv[0];
    if (t == 0 && ri[0] < NRES) swork[ri[0]] = FINF;
    __syncthreads();
  }

  // score
  for (int j = t; j < NRES; j += 256) {
    bool sb = (bn == batch[j]);
    bool sc = (cn == chain[j]);
    bool valid = sb && mn && (mask[j] != 0);
    int dr = rn - resi[j]; if (dr < 0) dr = -dr;
    bool within0 = (dr < 16) && sb && sc;
    float de = sdist[j];
    bool within = within0 || (de < cutoff);
    float s;
    if (!valid) s = FINF;
    else if (within) s = -10000.0f;
    else s = 3.0f * logf(fmaxf(de, 1e-6f));  // -(-3 log d)
    swork[j] = s;
  }
  __syncthreads();

  // extract K smallest
  for (int kk = 0; kk < KNB; ++kk) {
    float bv = FINF; int bi = NRES;
    for (int j = t; j < NRES; j += 256) {
      float v = swork[j];
      if (v < bv || (v == bv && j < bi)) { bv = v; bi = j; }
    }
    rv[t] = bv; ri[t] = bi;
    __syncthreads();
    for (int s = 128; s > 0; s >>= 1) {
      if (t < s) {
        if (rv[t + s] < rv[t] || (rv[t + s] == rv[t] && ri[t + s] < ri[t])) {
          rv[t] = rv[t + s]; ri[t] = ri[t + s];
        }
      }
      __syncthreads();
    }
    if (t == 0) {
      bool ok = (rv[0] < 1e30f) && mn && (ri[0] < NRES);
      nb[(size_t)n * KNB + kk] = ok ? ri[0] : -1;
      if (ri[0] < NRES) swork[ri[0]] = FINF;
    }
    __syncthreads();
  }
}

// ---------------------------------------------------------------------------
// Kernel 3: fused pair features -> proj(109->128) -> LN -> MLP(128->256->128)
//           -> bias head (128->8).  One residue (64 edge rows) at a time.
//           All four GEMMs run on v_wmma_f32_16x16x32_bf16 out of LDS.
// ---------------------------------------------------------------------------
#define PB_NPER 4
__global__ __launch_bounds__(256) void pair_bias_kernel(
    const float* __restrict__ pos, const float* __restrict__ Rw,
    const float* __restrict__ caw, const int* __restrict__ nb,
    const int* __restrict__ resi, const int* __restrict__ chain,
    const float* __restrict__ w_relpos, const float* __restrict__ w_dist,
    const float* __restrict__ w_dir, const float* __restrict__ w_rot,
    const float* __restrict__ w_vec, const float* __restrict__ ln_s,
    const float* __restrict__ ln_o, const float* __restrict__ w1g,
    const float* __restrict__ b1g, const float* __restrict__ w2g,
    const float* __restrict__ b2g, const float* __restrict__ wbg,
    float* __restrict__ biasout) {
  extern __shared__ char smem[];
  bf16_t* wcat = (bf16_t*)smem;               // 128x128 swizzled
  bf16_t* w1 = wcat + 128 * 128;              // 128x256 swizzled
  bf16_t* w2 = w1 + 128 * 256;                // 256x128 swizzled
  bf16_t* wb = w2 + 256 * 128;                // 128x16 swizzled
  float* b1 = (float*)(wb + 128 * 16);        // 256
  float* b2 = b1 + 256;                       // 128
  float* lns = b2 + 128;                      // 128
  float* lno = lns + 128;                     // 128
  bf16_t* featbf = (bf16_t*)(lno + 128);      // 64x128 row-major
  float* pairf = (float*)(featbf + 64 * 128); // 64x128
  bf16_t* h1bf = (bf16_t*)(pairf + 64 * 128); // 64x256 row-major
  bf16_t* h2bf = h1bf + 64 * 256;             // 64x128 row-major

  const int t = threadIdx.x;
  const int lane = t & 31;
  const int wvid = t >> 5;

  // stage weights (bf16, swizzled) once per block
  stage_swz(wcat, 128, 128, [&](int k, int n) -> float {
    if (k < 66) return w_relpos[k * 128 + n];
    if (k < 82) return w_dist[(k - 66) * 128 + n];
    if (k < 85) return w_dir[(k - 82) * 128 + n];
    if (k < 94) return w_rot[(k - 85) * 128 + n];
    if (k < 109) return w_vec[(k - 94) * 128 + n];
    return 0.f;
  });
  stage_swz(w1, 128, 256, [&](int k, int n) -> float { return w1g[k * 256 + n]; });
  stage_swz(w2, 256, 128, [&](int k, int n) -> float { return w2g[k * 128 + n]; });
  stage_swz(wb, 128, 16, [&](int k, int n) -> float {
    return (n < 8) ? wbg[k * 8 + n] : 0.f;
  });
  b1[t] = b1g[t];
  if (t < 128) { b2[t] = b2g[t]; lns[t] = ln_s[t]; lno[t] = ln_o[t]; }

  for (int it = 0; it < PB_NPER; ++it) {
    int n = blockIdx.x * PB_NPER + it;
    __syncthreads();

    // build 64 edge-feature rows (padded to 128)
    if (t < 64) {
      int k = t;
      int j = nb[(size_t)n * KNB + k];
      int jj = j < 0 ? 0 : j;
      bf16_t* fr = featbf + k * 128;
      for (int c = 0; c < 128; ++c) fr[c] = (bf16_t)0.f;
      int dr = resi[jj] - resi[n];
      dr = dr < -32 ? -32 : (dr > 32 ? 32 : dr);
      int rel = (chain[jj] == chain[n]) ? (dr + 32) : 65;
      fr[rel] = (bf16_t)1.f;
      float cax = caw[n * 3], cay = caw[n * 3 + 1], caz = caw[n * 3 + 2];
      float vx = caw[jj * 3] - cax, vy = caw[jj * 3 + 1] - cay, vz = caw[jj * 3 + 2] - caz;
      float d = sqrtf(vx * vx + vy * vy + vz * vz + 1e-12f);
#pragma unroll
      for (int i = 0; i < 16; ++i) {
        float cc = 22.f * (float)i / 15.f;
        float u = (d - cc) * (16.f / 22.f);
        fr[66 + i] = (bf16_t)__expf(-u * u);
      }
      const float* Rn = Rw + (size_t)n * 9;
      float dv0 = Rn[0] * vx + Rn[3] * vy + Rn[6] * vz;
      float dv1 = Rn[1] * vx + Rn[4] * vy + Rn[7] * vz;
      float dv2 = Rn[2] * vx + Rn[5] * vy + Rn[8] * vz;
      float dn = rsqrtf(dv0 * dv0 + dv1 * dv1 + dv2 * dv2 + 1e-12f);
      fr[82] = (bf16_t)(dv0 * dn);
      fr[83] = (bf16_t)(dv1 * dn);
      fr[84] = (bf16_t)(dv2 * dn);
      const float* Rj = Rw + (size_t)jj * 9;
#pragma unroll
      for (int r = 0; r < 3; ++r)
#pragma unroll
        for (int s = 0; s < 3; ++s) {
          float acc = Rn[0 + r] * Rj[0 + s] + Rn[3 + r] * Rj[3 + s] + Rn[6 + r] * Rj[6 + s];
          fr[85 + r * 3 + s] = (bf16_t)acc;
        }
#pragma unroll
      for (int a = 0; a < 5; ++a) {
        float ax = pos[(size_t)jj * 15 + a * 3 + 0] - cax;
        float ay = pos[(size_t)jj * 15 + a * 3 + 1] - cay;
        float az = pos[(size_t)jj * 15 + a * 3 + 2] - caz;
        fr[94 + a * 3 + 0] = (bf16_t)(Rn[0] * ax + Rn[3] * ay + Rn[6] * az);
        fr[94 + a * 3 + 1] = (bf16_t)(Rn[1] * ax + Rn[4] * ay + Rn[7] * az);
        fr[94 + a * 3 + 2] = (bf16_t)(Rn[2] * ax + Rn[5] * ay + Rn[8] * az);
      }
    }
    __syncthreads();

    // GEMM1: feat(64x128) @ wcat(128x128) -> pairf (f32)
    {
      v8f acc[4] = {};
#pragma unroll
      for (int ks = 0; ks < 128; ks += 32) {
        v16bf bf = ldb_frag_swz(wcat, 4, ks >> 5, wvid);
#pragma unroll
        for (int mt = 0; mt < 4; ++mt) {
          v16bf af = lda_frag(featbf + mt * 16 * 128, 128, ks);
          acc[mt] = wmma_bf16(af, bf, acc[mt]);
        }
      }
      int hh = lane >> 4, cl = lane & 15;
      int col = wvid * 16 + cl;
#pragma unroll
      for (int mt = 0; mt < 4; ++mt)
#pragma unroll
        for (int r = 0; r < 8; ++r)
          pairf[(mt * 16 + r + 8 * hh) * 128 + col] = acc[mt][r];
    }
    __syncthreads();

    // LayerNorm rows, write bf16 back into featbf (A for GEMM2)
    if (t < 64) {
      float* pr = pairf + t * 128;
      float sum = 0.f, sq = 0.f;
      for (int c = 0; c < 128; ++c) { float v = pr[c]; sum += v; sq += v * v; }
      float mean = sum * (1.f / 128.f);
      float var = sq * (1.f / 128.f) - mean * mean;
      float inv = rsqrtf(var + 1e-5f);
      bf16_t* fr = featbf + t * 128;
      for (int c = 0; c < 128; ++c)
        fr[c] = (bf16_t)((pr[c] - mean) * inv * lns[c] + lno[c]);
    }
    __syncthreads();

    // GEMM2: (64x128) @ w1(128x256) + b1 -> gelu -> h1bf
#pragma unroll
    for (int half = 0; half < 2; ++half) {
      int nt = wvid + half * 8;
      v8f acc[4] = {};
#pragma unroll
      for (int ks = 0; ks < 128; ks += 32) {
        v16bf bf = ldb_frag_swz(w1, 4, ks >> 5, nt);
#pragma unroll
        for (int mt = 0; mt < 4; ++mt) {
          v16bf af = lda_frag(featbf + mt * 16 * 128, 128, ks);
          acc[mt] = wmma_bf16(af, bf, acc[mt]);
        }
      }
      int hh = lane >> 4, cl = lane & 15;
      int col = nt * 16 + cl;
#pragma unroll
      for (int mt = 0; mt < 4; ++mt)
#pragma unroll
        for (int r = 0; r < 8; ++r) {
          float v = acc[mt][r] + b1[col];
          h1bf[(mt * 16 + r + 8 * hh) * 256 + col] = (bf16_t)gelu_f(v);
        }
    }
    __syncthreads();

    // GEMM3: (64x256) @ w2(256x128) + b2 -> h2bf
    {
      v8f acc[4] = {};
#pragma unroll
      for (int ks = 0; ks < 256; ks += 32) {
        v16bf bf = ldb_frag_swz(w2, 8, ks >> 5, wvid);
#pragma unroll
        for (int mt = 0; mt < 4; ++mt) {
          v16bf af = lda_frag(h1bf + mt * 16 * 256, 256, ks);
          acc[mt] = wmma_bf16(af, bf, acc[mt]);
        }
      }
      int hh = lane >> 4, cl = lane & 15;
      int col = wvid * 16 + cl;
#pragma unroll
      for (int mt = 0; mt < 4; ++mt)
#pragma unroll
        for (int r = 0; r < 8; ++r)
          h2bf[(mt * 16 + r + 8 * hh) * 128 + col] = (bf16_t)(acc[mt][r] + b2[col]);
    }
    __syncthreads();

    // GEMM4: (64x128) @ wb(128x16, cols 8..15 zero) -> bias out
    if (wvid < 4) {
      v8f acc = {};
#pragma unroll
      for (int ks = 0; ks < 128; ks += 32) {
        v16bf bf = ldb_frag_swz(wb, 4, ks >> 5, 0);
        v16bf af = lda_frag(h2bf + wvid * 16 * 128, 128, ks);
        acc = wmma_bf16(af, bf, acc);
      }
      int hh = lane >> 4, cl = lane & 15;
      if (cl < 8) {
#pragma unroll
        for (int r = 0; r < 8; ++r) {
          int row = wvid * 16 + r + 8 * hh;
          biasout[((size_t)n * KNB + row) * NHEAD + cl] = acc[r];
        }
      }
    }
  }
}

// ---------------------------------------------------------------------------
// Kernel 4: generic bf16-WMMA GEMM: out = res + mask*(act(A@W + b))
// Block = 64 rows, loops 128-col chunks. A fp32 (lda), W fp32 (ldw=N).
// ---------------------------------------------------------------------------
__global__ __launch_bounds__(256) void gemm_bf16_kernel(
    const float* __restrict__ A, int lda, const float* __restrict__ W, int ldw,
    const float* __restrict__ bias, const float* __restrict__ resid, int ldres,
    const unsigned char* __restrict__ mrow, float* __restrict__ out, int ldout,
    int ocol0, int Nout, int Kdim, int dogelu) {
  extern __shared__ char smem[];
  const int Kpad = (Kdim + 63) & ~63;
  bf16_t* Abf = (bf16_t*)smem;        // 64 x Kpad row-major
  bf16_t* Wbf = Abf + 64 * Kpad;      // Kpad x 128 swizzled
  const int t = threadIdx.x;
  const int lane = t & 31;
  const int wvid = t >> 5;
  const int m0 = blockIdx.x * 64;

  for (int idx = t; idx < 64 * Kpad; idx += 256) {
    int r = idx / Kpad, c = idx - r * Kpad;
    float v = (c < Kdim) ? A[(size_t)(m0 + r) * lda + c] : 0.f;
    Abf[idx] = (bf16_t)v;
  }

  for (int nc = 0; nc < Nout; nc += 128) {
    __syncthreads();
    stage_swz(Wbf, Kpad, 128, [&](int k, int n) -> float {
      return (k < Kdim) ? W[(size_t)k * ldw + nc + n] : 0.f;
    });
    if (nc + 128 < Nout && t < Kdim)
      __builtin_prefetch(&W[(size_t)t * ldw + nc + 128], 0, 0);
    __syncthreads();

    v8f acc[4] = {};
    for (int ks = 0; ks < Kpad; ks += 32) {
      v16bf bf = ldb_frag_swz(Wbf, Kpad >> 5, ks >> 5, wvid);
#pragma unroll
      for (int mt = 0; mt < 4; ++mt) {
        v16bf af = lda_frag(Abf + mt * 16 * Kpad, Kpad, ks);
        acc[mt] = wmma_bf16(af, bf, acc[mt]);
      }
    }
    int hh = lane >> 4, cl = lane & 15;
    int col = nc + wvid * 16 + cl;
#pragma unroll
    for (int mt = 0; mt < 4; ++mt)
#pragma unroll
      for (int r = 0; r < 8; ++r) {
        int row = m0 + mt * 16 + r + 8 * hh;
        float v = acc[mt][r];
        if (bias) v += bias[col];
        if (dogelu) v = gelu_f(v);
        if (resid) {
          float mm = mrow ? (mrow[row] ? 1.f : 0.f) : 1.f;
          v = resid[(size_t)row * ldres + col] + v * mm;
        }
        out[(size_t)row * ldout + ocol0 + col] = v;
      }
  }
}

// ---------------------------------------------------------------------------
// Kernel 5: layernorm (one wave per row)
// ---------------------------------------------------------------------------
__global__ __launch_bounds__(256) void layernorm_kernel(
    const float* __restrict__ x, const float* __restrict__ s,
    const float* __restrict__ o, float* __restrict__ y, int C) {
  int row = blockIdx.x * 8 + (threadIdx.x >> 5);
  int lane = threadIdx.x & 31;
  const float* xr = x + (size_t)row * C;
  float sum = 0.f, sq = 0.f;
  for (int c = lane; c < C; c += 32) { float v = xr[c]; sum += v; sq += v * v; }
#pragma unroll
  for (int m = 16; m > 0; m >>= 1) {
    sum += __shfl_xor(sum, m, 32);
    sq += __shfl_xor(sq, m, 32);
  }
  float mean = sum / (float)C;
  float var = sq / (float)C - mean * mean;
  float inv = rsqrtf(var + 1e-5f);
  float* yr = y + (size_t)row * C;
  for (int c = lane; c < C; c += 32) yr[c] = (xr[c] - mean) * inv * s[c] + o[c];
}

// ---------------------------------------------------------------------------
// Kernel 6: neighborhood attention softmax (block per residue).
// K/V rows are gathered into LDS with CDNA5 async global->LDS copies.
// ---------------------------------------------------------------------------
__global__ __launch_bounds__(256) void attention_kernel(
    const float* __restrict__ qkv, const int* __restrict__ nb,
    const float* __restrict__ biasw, const unsigned char* __restrict__ mask,
    float* __restrict__ outp) {
  extern __shared__ char smem[];
  float* kbuf = (float*)smem;           // 64 x 256
  float* vbuf = kbuf + 64 * 256;        // 64 x 256
  float* qs = vbuf + 64 * 256;          // 256
  float* slog = qs + 256;               // 64 x 8
  float* sattn = slog + 512;            // 64 x 8
  int* jidx = (int*)(sattn + 512);      // 64
  float* pm = (float*)(jidx + 64);      // 64

  int n = blockIdx.x;
  int t = threadIdx.x;
  int lane = t & 31, wvid = t >> 5;
  if (t < 64) {
    int j = nb[(size_t)n * KNB + t];
    int jj = j < 0 ? 0 : j;
    jidx[t] = jj;
    pm[t] = (j >= 0 && mask[n] && mask[jj]) ? 1.f : 0.f;
  }
  qs[t] = qkv[(size_t)n * 768 + t];
  __syncthreads();
  // async gather of K/V neighbor rows into LDS (per-lane 16B copies)
  for (int r = wvid; r < 64; r += 8) {
    const float* ksrc = qkv + (size_t)jidx[r] * 768 + 256;
    float* kdst = kbuf + r * 256;
    async_copy_b128(kdst + lane * 4, ksrc + lane * 4);
    async_copy_b128(kdst + 128 + lane * 4, ksrc + 128 + lane * 4);
    const float* vsrc = ksrc + 256;
    float* vdst = vbuf + r * 256;
    async_copy_b128(vdst + lane * 4, vsrc + lane * 4);
    async_copy_b128(vdst + 128 + lane * 4, vsrc + 128 + lane * 4);
  }
  async_wait0();
  __syncthreads();
  const float isq = 0.17677669529663687f;  // 1/sqrt(32)
  for (int p = t; p < 512; p += 256) {
    int k = p & 63, h = p >> 6;
    float d = 0.f;
#pragma unroll
    for (int d0 = 0; d0 < 32; ++d0) d += qs[h * 32 + d0] * kbuf[k * 256 + h * 32 + d0];
    float l = d * isq + biasw[((size_t)n * KNB + k) * NHEAD + h];
    if (pm[k] == 0.f) l = -1e9f;
    slog[k * 8 + h] = l;
  }
  __syncthreads();
  if (t < 8) {
    float mx = -3e38f;
    for (int k = 0; k < 64; ++k) mx = fmaxf(mx, slog[k * 8 + t]);
    float s = 0.f;
    for (int k = 0; k < 64; ++k) {
      float e = __expf(slog[k * 8 + t] - mx);
      sattn[k * 8 + t] = e;
      s += e;
    }
    float inv = 1.f / s;
    for (int k = 0; k < 64; ++k) sattn[k * 8 + t] *= inv;
  }
  __syncthreads();
  {
    int h = t >> 5, d0 = t & 31;
    float o = 0.f;
    for (int k = 0; k < 64; ++k) o += sattn[k * 8 + h] * vbuf[k * 256 + h * 32 + d0];
    outp[(size_t)n * DMODEL + h * 32 + d0] = o;
  }
}

__global__ __launch_bounds__(256) void mul_kernel(const float* __restrict__ a,
                                                  const float* __restrict__ b,
                                                  float* __restrict__ c, int nElem) {
  int i = blockIdx.x * blockDim.x + threadIdx.x;
  if (i < nElem) c[i] = a[i] * b[i];
}

// ---------------------------------------------------------------------------
// Host orchestration
// ---------------------------------------------------------------------------
extern "C" void kernel_launch(void* const* d_in, const int* in_sizes, int n_in,
                              void* d_out, int out_size, void* d_ws, size_t ws_size,
                              hipStream_t stream) {
  (void)in_sizes; (void)n_in; (void)out_size; (void)ws_size;
  const float* features = (const float*)d_in[0];
  const float* pos = (const float*)d_in[1];
  const int* resi = (const int*)d_in[2];
  const int* chain = (const int*)d_in[3];
  const int* batch = (const int*)d_in[4];
  const unsigned char* mask = (const unsigned char*)d_in[5];
  const float* w_relpos = (const float*)d_in[6];
  const float* w_dist = (const float*)d_in[7];
  const float* w_dir = (const float*)d_in[8];
  const float* w_rot = (const float*)d_in[9];
  const float* w_vec = (const float*)d_in[10];
  const float* ln_pair_s = (const float*)d_in[11];
  const float* ln_pair_o = (const float*)d_in[12];
  const float* pmlp_w1 = (const float*)d_in[13];
  const float* pmlp_b1 = (const float*)d_in[14];
  const float* pmlp_w2 = (const float*)d_in[15];
  const float* pmlp_b2 = (const float*)d_in[16];
  const float* ln_attn_s = (const float*)d_in[17];
  const float* ln_attn_o = (const float*)d_in[18];
  const float* wq = (const float*)d_in[19];
  const float* wk = (const float*)d_in[20];
  const float* wvw = (const float*)d_in[21];
  const float* wbp = (const float*)d_in[22];
  const float* wo = (const float*)d_in[23];
  const float* ln_upd_s = (const float*)d_in[24];
  const float* ln_upd_o = (const float*)d_in[25];
  const float* pos_w1 = (const float*)d_in[26];
  const float* pos_b1 = (const float*)d_in[27];
  const float* pos_w2 = (const float*)d_in[28];
  const float* pos_b2 = (const float*)d_in[29];
  const float* w_up = (const float*)d_in[30];
  const float* w_gate = (const float*)d_in[31];
  const float* w_out = (const float*)d_in[32];
  const float* b_out = (const float*)d_in[33];
  float* out = (float*)d_out;

  // workspace carve
  char* wsp = (char*)d_ws;
  auto alloc = [&](size_t bytes) {
    char* p = wsp;
    wsp += (bytes + 255) & ~(size_t)255;
    return p;
  };
  float* R = (float*)alloc((size_t)NRES * 9 * 4);
  float* ca = (float*)alloc((size_t)NRES * 3 * 4);
  float* lp = (float*)alloc((size_t)NRES * 16 * 4);
  int* nb = (int*)alloc((size_t)NRES * KNB * 4);
  float* biasb = (float*)alloc((size_t)NRES * KNB * NHEAD * 4);
  float* xln = (float*)alloc((size_t)NRES * DMODEL * 4);
  float* qkv = (float*)alloc((size_t)NRES * 768 * 4);
  float* attnout = (float*)alloc((size_t)NRES * DMODEL * 4);
  float* feat2 = (float*)alloc((size_t)NRES * DMODEL * 4);
  float* yln = (float*)alloc((size_t)NRES * DMODEL * 4);
  float* posh = (float*)alloc((size_t)NRES * 512 * 4);
  float* yv = (float*)alloc((size_t)NRES * DMODEL * 4);
  float* up = (float*)alloc((size_t)NRES * 512 * 4);
  float* gate = (float*)alloc((size_t)NRES * 512 * 4);
  float* gu = (float*)alloc((size_t)NRES * 512 * 4);

  auto gemm = [&](const float* A, int lda, const float* W, int ldw,
                  const float* bias, const float* res, int ldres,
                  const unsigned char* mk, float* o, int ldo, int oc0, int Nout,
                  int Kdim, int dogelu) {
    int Kpad = (Kdim + 63) & ~63;
    size_t sm = (size_t)64 * Kpad * 2 + (size_t)Kpad * 128 * 2;
    gemm_bf16_kernel<<<NRES / 64, 256, sm, stream>>>(A, lda, W, ldw, bias, res,
                                                     ldres, mk, o, ldo, oc0,
                                                     Nout, Kdim, dogelu);
  };

  frames_kernel<<<NRES / 256, 256, 0, stream>>>(pos, R, ca, lp);
  neighbours_kernel<<<NRES, 256, 0, stream>>>(ca, resi, chain, batch, mask, nb);

  size_t smPB = (size_t)(128 * 128 + 128 * 256 + 256 * 128 + 128 * 16) * 2 +
                (size_t)(256 + 128 * 3) * 4 +
                (size_t)(64 * 128) * 2 + (size_t)(64 * 128) * 4 +
                (size_t)(64 * 256) * 2 + (size_t)(64 * 128) * 2;
  pair_bias_kernel<<<NRES / PB_NPER, 256, smPB, stream>>>(
      pos, R, ca, nb, resi, chain, w_relpos, w_dist, w_dir, w_rot, w_vec,
      ln_pair_s, ln_pair_o, pmlp_w1, pmlp_b1, pmlp_w2, pmlp_b2, wbp, biasb);

  layernorm_kernel<<<NRES / 8, 256, 0, stream>>>(features, ln_attn_s, ln_attn_o, xln, DMODEL);
  gemm(xln, 256, wq, 256, nullptr, nullptr, 0, nullptr, qkv, 768, 0, 256, 256, 0);
  gemm(xln, 256, wk, 256, nullptr, nullptr, 0, nullptr, qkv, 768, 256, 256, 256, 0);
  gemm(xln, 256, wvw, 256, nullptr, nullptr, 0, nullptr, qkv, 768, 512, 256, 256, 0);

  size_t smAT = (size_t)(64 * 256 * 2 + 256 + 512 + 512) * 4 + 64 * 4 + 64 * 4;
  attention_kernel<<<NRES, 256, smAT, stream>>>(qkv, nb, biasb, mask, attnout);

  gemm(attnout, 256, wo, 256, nullptr, features, 256, mask, feat2, 256, 0, 256, 256, 0);
  layernorm_kernel<<<NRES / 8, 256, 0, stream>>>(feat2, ln_upd_s, ln_upd_o, yln, DMODEL);
  gemm(lp, 16, pos_w1, 512, pos_b1, nullptr, 0, nullptr, posh, 512, 0, 512, 15, 1);
  gemm(posh, 512, pos_w2, 256, pos_b2, yln, 256, nullptr, yv, 256, 0, 256, 512, 0);
  gemm(yv, 256, w_up, 512, nullptr, nullptr, 0, nullptr, up, 512, 0, 512, 256, 0);
  gemm(yv, 256, w_gate, 512, nullptr, nullptr, 0, nullptr, gate, 512, 0, 512, 256, 1);
  mul_kernel<<<(NRES * 512) / 256, 256, 0, stream>>>(gate, up, gu, NRES * 512);
  gemm(gu, 512, w_out, 256, b_out, feat2, 256, mask, out, 256, 0, 256, 512, 0);
}
